// SmyrfAttention_15788299780115
// MI455X (gfx1250) — matrix operand order repaired and checked by reading, not compiled, not run on hardware
//
#include <hip/hip_runtime.h>
#include <math.h>

#define BATCH 16
#define SEQ   4096
#define DIM   64
#define NH    8
#define CL    128
#define NC    (SEQ / CL)

typedef __attribute__((ext_vector_type(16))) _Float16 v16h;
typedef __attribute__((ext_vector_type(8)))  float    v8f;

// ---------- WMMA helpers ----------
// A-fragment (16x32 f16): lane m=l&15 holds row (row_base+m); element e holds
// K = col_base + 16*(e/8) + 8*(lane/16) + (e&7)  -> two 16B LDS loads.
__device__ __forceinline__ v16h ld_fragA(const _Float16* lds, int stride,
                                         int row_base, int col_base, int lane) {
  const int m = lane & 15, hl = (lane >> 4) & 1;
  union { v16h v; uint4 q[2]; } f;
  const _Float16* p = lds + (row_base + m) * stride + col_base + hl * 8;
  f.q[0] = *(const uint4*)(p);
  f.q[1] = *(const uint4*)(p + 16);
  return f.v;
}

// B-fragment (32x16 f16): lane l holds column N = n_base + (l&15); element e holds
// K = k_base + 16*(e/8) + 8*(lane/16) + (e&7).
__device__ __forceinline__ v16h ld_fragB(const _Float16* lds, int stride,
                                         int k_base, int n_base, int lane) {
  const int n = lane & 15, hl = (lane >> 4) & 1;
  v16h f;
  #pragma unroll
  for (int e = 0; e < 16; e++)
    f[e] = lds[(k_base + 8 * hl + (e & 7) + 16 * (e >> 3)) * stride + n_base + n];
  return f;
}

__device__ __forceinline__ v8f wmma16(v16h a, v16h b, v8f c) {
  return __builtin_amdgcn_wmma_f32_16x16x32_f16(false, a, false, b,
                                                (short)0, c, false, false);
}

// ---------- Phase 1: squared norms ----------
__global__ void norms_kernel(const float* __restrict__ q, const float* __restrict__ k,
                             float* __restrict__ qn2, float* __restrict__ kn2) {
  int i = blockIdx.x * blockDim.x + threadIdx.x;     // b*SEQ + s
  const float4* qp = (const float4*)(q + (size_t)i * DIM);
  const float4* kp = (const float4*)(k + (size_t)i * DIM);
  float sq = 0.f, sk = 0.f;
  #pragma unroll
  for (int j = 0; j < DIM / 4; j++) {
    float4 a = qp[j], b = kp[j];
    sq += a.x * a.x + a.y * a.y + a.z * a.z + a.w * a.w;
    sk += b.x * b.x + b.y * b.y + b.z * b.z + b.w * b.w;
  }
  qn2[i] = sq; kn2[i] = sk;
}

// ---------- Phase 2: per-batch max norm^2 ----------
__global__ void maxred_kernel(const float* __restrict__ qn2, const float* __restrict__ kn2,
                              float* __restrict__ M2) {
  int b = blockIdx.x, tid = threadIdx.x;
  float mx = 0.f;
  for (int s = tid; s < SEQ; s += 256) {
    mx = fmaxf(mx, qn2[b * SEQ + s]);
    mx = fmaxf(mx, kn2[b * SEQ + s]);
  }
  __shared__ float red[256];
  red[tid] = mx; __syncthreads();
  for (int off = 128; off > 0; off >>= 1) {
    if (tid < off) red[tid] = fmaxf(red[tid], red[tid + off]);
    __syncthreads();
  }
  if (tid == 0) M2[b] = red[0];
}

// ---------- Phase 3: E2LSH hashes with XBOX+ augmentation ----------
__global__ void hash_kernel(const float* __restrict__ q, const float* __restrict__ k,
                            const float* __restrict__ alpha, const float* __restrict__ beta,
                            const float* __restrict__ qn2, const float* __restrict__ kn2,
                            const float* __restrict__ M2,
                            float* __restrict__ qhash, float* __restrict__ khash) {
  int i = blockIdx.x * blockDim.x + threadIdx.x;     // b*SEQ + s
  int b = i >> 12, s = i & (SEQ - 1);
  float aq[NH], ak[NH];
  #pragma unroll
  for (int h = 0; h < NH; h++) { aq[h] = beta[h]; ak[h] = beta[h]; }
  const float* qr = q + (size_t)i * DIM;
  const float* kr = k + (size_t)i * DIM;
  for (int d = 0; d < DIM; d++) {
    float qd = qr[d], kd = kr[d];
    #pragma unroll
    for (int h = 0; h < NH; h++) {
      float al = alpha[d * NH + h];
      aq[h] += qd * al; ak[h] += kd * al;
    }
  }
  float m2 = M2[b];
  float qe = sqrtf(fmaxf(m2 - qn2[i], 0.f));
  float ke = sqrtf(fmaxf(m2 - kn2[i], 0.f));
  #pragma unroll
  for (int h = 0; h < NH; h++) {
    aq[h] += qe * alpha[DIM * NH + h];          // Q ext at slot dim
    ak[h] += ke * alpha[(DIM + 1) * NH + h];    // K ext at slot dim+1
    qhash[((size_t)h * BATCH + b) * SEQ + s] = aq[h];
    khash[((size_t)h * BATCH + b) * SEQ + s] = ak[h];
  }
}

// ---------- Phase 4: bitonic argsort per (h,b), 4096 elems in LDS ----------
__global__ __launch_bounds__(1024) void sort_kernel(const float* __restrict__ qhash,
                                                    const float* __restrict__ khash,
                                                    int* __restrict__ qpos,
                                                    int* __restrict__ kpos) {
  const int hb  = blockIdx.x;
  const float* src = (blockIdx.y == 0) ? qhash : khash;
  int*         dst = (blockIdx.y == 0) ? qpos  : kpos;
  __shared__ float skey[SEQ];
  __shared__ int   sidx[SEQ];
  const int tid = threadIdx.x;
  for (int i = tid; i < SEQ; i += 1024) { skey[i] = src[(size_t)hb * SEQ + i]; sidx[i] = i; }
  __syncthreads();
  for (int kk = 2; kk <= SEQ; kk <<= 1) {
    for (int j = kk >> 1; j > 0; j >>= 1) {
      for (int t = tid; t < SEQ / 2; t += 1024) {
        int i = ((t & ~(j - 1)) << 1) | (t & (j - 1));
        int p = i + j;
        bool up = ((i & kk) == 0);
        float ka = skey[i], kb = skey[p];
        int   ia = sidx[i], ib = sidx[p];
        bool agtb = (ka > kb) || (ka == kb && ia > ib);   // total order (stable)
        if (agtb == up) { skey[i] = kb; skey[p] = ka; sidx[i] = ib; sidx[p] = ia; }
      }
      __syncthreads();
    }
  }
  for (int i = tid; i < SEQ; i += 1024) dst[(size_t)hb * SEQ + i] = sidx[i];
}

// ---------- Phase 5: per-cluster QK^T via WMMA -> row LSE, scattered ----------
__global__ __launch_bounds__(256) void lse_kernel(const float* __restrict__ q,
                                                  const float* __restrict__ k,
                                                  const int* __restrict__ qpos,
                                                  const int* __restrict__ kpos,
                                                  float* __restrict__ lse_out) {
  const int c = blockIdx.x, b = blockIdx.y, h = blockIdx.z;
  const int tid = threadIdx.x, lane = tid & 31, w = tid >> 5;
  const int m = lane & 15, hl = (lane >> 4) & 1;
  __shared__ _Float16 sq[CL * 72];        // Q row-major [128][64] pad
  __shared__ _Float16 skT[DIM * 136];     // K transposed [64][128] pad
  __shared__ int qidx[CL], kidx[CL];
  const size_t hb = (size_t)h * BATCH + b;
  if (tid < CL) {
    qidx[tid] = qpos[hb * SEQ + c * CL + tid];
    kidx[tid] = kpos[hb * SEQ + c * CL + tid];
  }
  __syncthreads();
  for (int i = tid; i < CL * DIM; i += 256) {
    int r = i >> 6, d = i & 63;
    sq[r * 72 + d]   = (_Float16)q[((size_t)b * SEQ + qidx[r]) * DIM + d];
    skT[d * 136 + r] = (_Float16)k[((size_t)b * SEQ + kidx[r]) * DIM + d];
  }
  __syncthreads();
  v8f acc[8] = {};
  #pragma unroll
  for (int chunk = 0; chunk < DIM; chunk += 32) {
    v16h a = ld_fragA(sq, 72, 16 * w, chunk, lane);
    #pragma unroll
    for (int kt = 0; kt < 8; kt++) {
      v16h bf = ld_fragB(skT, 136, chunk, 16 * kt, lane);
      acc[kt] = wmma16(a, bf, acc[kt]);
    }
  }
  // row stats: row = 16w + 8*hl + r spread across 16 lanes (n) x 8 tiles
  #pragma unroll
  for (int r = 0; r < 8; r++) {
    float mx = -3.0e38f;
    #pragma unroll
    for (int kt = 0; kt < 8; kt++) mx = fmaxf(mx, acc[kt][r]);
    for (int msk = 8; msk >= 1; msk >>= 1) mx = fmaxf(mx, __shfl_xor(mx, msk, 32));
    float sum = 0.f;
    #pragma unroll
    for (int kt = 0; kt < 8; kt++) sum += __expf(acc[kt][r] - mx);
    for (int msk = 8; msk >= 1; msk >>= 1) sum += __shfl_xor(sum, msk, 32);
    if (m == 0) {
      int row = 16 * w + 8 * hl + r;
      lse_out[hb * SEQ + qidx[row]] = mx + __logf(sum);
    }
  }
}

// ---------- Phase 6: softmax over hash rounds ----------
__global__ void prob_kernel(const float* __restrict__ lse, float* __restrict__ prob) {
  int i = blockIdx.x * blockDim.x + threadIdx.x;     // b*SEQ + s
  float v[NH], mx = -3.0e38f;
  #pragma unroll
  for (int h = 0; h < NH; h++) { v[h] = lse[(size_t)h * BATCH * SEQ + i]; mx = fmaxf(mx, v[h]); }
  float sum = 0.f;
  #pragma unroll
  for (int h = 0; h < NH; h++) sum += __expf(v[h] - mx);
  float lt = mx + __logf(sum);
  #pragma unroll
  for (int h = 0; h < NH; h++) prob[(size_t)h * BATCH * SEQ + i] = __expf(v[h] - lt);
}

// ---------- Phase 7: full attention, weight by prob, atomic scatter ----------
__global__ __launch_bounds__(256) void attn_kernel(const float* __restrict__ q,
                                                   const float* __restrict__ k,
                                                   const float* __restrict__ v,
                                                   const int* __restrict__ qpos,
                                                   const int* __restrict__ kpos,
                                                   const float* __restrict__ lse,
                                                   const float* __restrict__ prob,
                                                   float* __restrict__ out) {
  const int c = blockIdx.x, b = blockIdx.y, h = blockIdx.z;
  const int tid = threadIdx.x, lane = tid & 31, w = tid >> 5;
  const int m = lane & 15, hl = (lane >> 4) & 1;
  // qkbuf holds sq(128*72) + skT(64*136) during QK; reused as P[128][136] f16
  __shared__ _Float16 qkbuf[CL * 72 + DIM * 136];
  __shared__ _Float16 sv[CL * 72];        // V row-major [128][64] pad
  __shared__ int qidx[CL], kidx[CL];
  __shared__ float lse_l[CL], pw[CL];
  _Float16* sq  = qkbuf;
  _Float16* skT = qkbuf + CL * 72;
  const size_t hb = (size_t)h * BATCH + b;
  if (tid < CL) {
    int qi = qpos[hb * SEQ + c * CL + tid];
    int ki = kpos[hb * SEQ + c * CL + tid];
    qidx[tid] = qi; kidx[tid] = ki;
    lse_l[tid] = lse[hb * SEQ + qi];
    pw[tid]    = prob[hb * SEQ + qi];
  }
  __syncthreads();
  for (int i = tid; i < CL * DIM; i += 256) {
    int r = i >> 6, d = i & 63;
    sq[r * 72 + d]   = (_Float16)q[((size_t)b * SEQ + qidx[r]) * DIM + d];
    skT[d * 136 + r] = (_Float16)k[((size_t)b * SEQ + kidx[r]) * DIM + d];
    sv[r * 72 + d]   = (_Float16)v[((size_t)b * SEQ + kidx[r]) * DIM + d];
  }
  __syncthreads();
  v8f acc[8] = {};
  #pragma unroll
  for (int chunk = 0; chunk < DIM; chunk += 32) {
    v16h a = ld_fragA(sq, 72, 16 * w, chunk, lane);
    #pragma unroll
    for (int kt = 0; kt < 8; kt++) {
      v16h bf = ld_fragB(skT, 136, chunk, 16 * kt, lane);
      acc[kt] = wmma16(a, bf, acc[kt]);
    }
  }
  __syncthreads();   // sq/skT dead; qkbuf becomes P
  #pragma unroll
  for (int r = 0; r < 8; r++) {
    int row = 16 * w + 8 * hl + r;
    float l = lse_l[row];
    #pragma unroll
    for (int kt = 0; kt < 8; kt++)
      qkbuf[row * 136 + 16 * kt + m] = (_Float16)__expf(acc[kt][r] - l);
  }
  __syncthreads();
  v8f oacc[4] = {};
  #pragma unroll
  for (int chunk = 0; chunk < CL; chunk += 32) {
    v16h a = ld_fragA(qkbuf, 136, 16 * w, chunk, lane);
    #pragma unroll
    for (int vt = 0; vt < 4; vt++) {
      v16h bf = ld_fragB(sv, 72, chunk, 16 * vt, lane);
      oacc[vt] = wmma16(a, bf, oacc[vt]);
    }
  }
  #pragma unroll
  for (int r = 0; r < 8; r++) {
    int row = 16 * w + 8 * hl + r;
    int orig = qidx[row];
    float wgt = pw[row];
    float* orow = out + ((size_t)b * SEQ + orig) * DIM + m;
    #pragma unroll
    for (int vt = 0; vt < 4; vt++)
      atomicAdd(orow + 16 * vt, oacc[vt][r] * wgt);
  }
}

extern "C" void kernel_launch(void* const* d_in, const int* in_sizes, int n_in,
                              void* d_out, int out_size, void* d_ws, size_t ws_size,
                              hipStream_t stream) {
  (void)in_sizes; (void)n_in; (void)ws_size;
  const float* q     = (const float*)d_in[0];
  const float* k     = (const float*)d_in[1];
  const float* v     = (const float*)d_in[2];
  const float* alpha = (const float*)d_in[3];   // [66][8]
  const float* beta  = (const float*)d_in[4];   // [8]
  float* out = (float*)d_out;

  char* ws = (char*)d_ws;
  size_t off = 0;
  auto alloc = [&](size_t bytes) { void* p = ws + off; off += (bytes + 255) & ~(size_t)255; return p; };
  float* qn2   = (float*)alloc((size_t)BATCH * SEQ * 4);
  float* kn2   = (float*)alloc((size_t)BATCH * SEQ * 4);
  float* M2    = (float*)alloc(BATCH * 4);
  float* qhash = (float*)alloc((size_t)NH * BATCH * SEQ * 4);
  float* khash = (float*)alloc((size_t)NH * BATCH * SEQ * 4);
  int*   qpos  = (int*)  alloc((size_t)NH * BATCH * SEQ * 4);
  int*   kpos  = (int*)  alloc((size_t)NH * BATCH * SEQ * 4);
  float* lseb  = (float*)alloc((size_t)NH * BATCH * SEQ * 4);
  float* probb = (float*)alloc((size_t)NH * BATCH * SEQ * 4);

  hipMemsetAsync(out, 0, (size_t)out_size * sizeof(float), stream);
  norms_kernel <<<(BATCH * SEQ) / 256, 256, 0, stream>>>(q, k, qn2, kn2);
  maxred_kernel<<<BATCH, 256, 0, stream>>>(qn2, kn2, M2);
  hash_kernel  <<<(BATCH * SEQ) / 256, 256, 0, stream>>>(q, k, alpha, beta, qn2, kn2, M2, qhash, khash);
  sort_kernel  <<<dim3(NH * BATCH, 2), 1024, 0, stream>>>(qhash, khash, qpos, kpos);
  lse_kernel   <<<dim3(NC, BATCH, NH), 256, 0, stream>>>(q, k, qpos, kpos, lseb);
  prob_kernel  <<<(BATCH * SEQ) / 256, 256, 0, stream>>>(lseb, probb);
  attn_kernel  <<<dim3(NC, BATCH, NH), 256, 0, stream>>>(q, k, v, qpos, kpos, lseb, probb, out);
}
